// SelfAttention_30004641529897
// MI455X (gfx1250) — compile-verified, matrix operand order checked by reading, not run on hardware
//
#include <hip/hip_runtime.h>

// ---- problem constants -------------------------------------------------
#define BATCH 8
#define TQ    4096          // queries
#define SK    2048          // keys (T/2 after maxpool)
#define CIN   512
#define DKEY  64
#define DVAL  256
#define MPROJ 384           // 64(q)+64(k)+256(v) concatenated projection rows

typedef __bf16 bf16_t;
typedef __attribute__((ext_vector_type(16))) __bf16 v16bf;
typedef __attribute__((ext_vector_type(8)))  __bf16 v8bf;
typedef __attribute__((ext_vector_type(8)))  float  v8f;

// ---- WMMA helpers ------------------------------------------------------
__device__ __forceinline__ v8f wmma_bf16(v16bf a, v16bf b, v8f c) {
  // D(16x16,f32) = A(16x32,bf16) * B(32x16,bf16) + C
  return __builtin_amdgcn_wmma_f32_16x16x32_bf16(false, a, false, b,
                                                 (short)0, c, false, false);
}

// A-fragment (16x32, row-major source, ld elements per row).
// Lanes 0-15: row m, K = k0+{0..7, 16..23}; lanes 16-31: K = k0+{8..15, 24..31}.
// Same loader serves B-fragments when given B^T (rows = N).
__device__ __forceinline__ v16bf load_frag(const bf16_t* __restrict__ base,
                                           int ld, int k0, int lane) {
  const int m  = lane & 15;
  const int kh = (lane >> 4) << 3;
  const bf16_t* p = base + (size_t)m * ld + k0 + kh;
  union { v16bf v; v8bf h[2]; } u;
  u.h[0] = *(const v8bf*)(p);
  u.h[1] = *(const v8bf*)(p + 16);
  return u.v;
}

__device__ __forceinline__ v8bf pack8(v8f a) {
  v8bf p;
#pragma unroll
  for (int r = 0; r < 8; ++r) p[r] = (bf16_t)a[r];
  return p;
}

// ---- Kernel A: weights -> bf16 (Wq|Wk|Wv concatenated) -----------------
__global__ void prep_weights(const float* __restrict__ Wq, const float* __restrict__ Wk,
                             const float* __restrict__ Wv, const float* __restrict__ Wf,
                             bf16_t* __restrict__ Wcat, bf16_t* __restrict__ WfB) {
  const int i = blockIdx.x * blockDim.x + threadIdx.x;
  const int NQ = DKEY * CIN;            // 32768
  const int NV = DVAL * CIN;            // 131072
  const int NF = CIN * DVAL;            // 131072
  if (i < NQ)                       Wcat[i] = (bf16_t)Wq[i];
  else if (i < 2 * NQ)              Wcat[i] = (bf16_t)Wk[i - NQ];
  else if (i < 2 * NQ + NV)         Wcat[i] = (bf16_t)Wv[i - 2 * NQ];
  else if (i < 2 * NQ + NV + NF)    WfB[i - (2 * NQ + NV)] = (bf16_t)Wf[i - (2 * NQ + NV)];
}

// ---- Kernel B: x[b,c,t] f32 -> xT[b,t,c] bf16 (LDS tiled transpose) ----
__global__ void transpose_x(const float* __restrict__ x, bf16_t* __restrict__ xT) {
  __shared__ float tile[32][33];
  const int t0 = blockIdx.x * 32, c0 = blockIdx.y * 32, b = blockIdx.z;
  for (int i = threadIdx.y; i < 32; i += 8)
    tile[i][threadIdx.x] = x[((size_t)b * CIN + c0 + i) * TQ + t0 + threadIdx.x];
  __syncthreads();
  for (int i = threadIdx.y; i < 32; i += 8)
    xT[((size_t)b * TQ + t0 + i) * CIN + c0 + threadIdx.x] =
        (bf16_t)tile[threadIdx.x][i];
}

// ---- Kernel C: fused projections + maxpool -----------------------------
// P[ch,t] = sum_c Wcat[ch,c] * xT[t,c]   (WMMA, K=512)
// rows 0-63   -> qT[b,t,c]                (no pool)
// rows 64-127 -> kT[b,s,c]  s=t/2         (pool via shfl_xor lane pairs)
// rows 128-383-> v [b,cv,s]               (pool, row-major for A-frags later)
__global__ void proj_kernel(const bf16_t* __restrict__ Wcat, const bf16_t* __restrict__ xT,
                            bf16_t* __restrict__ qT, bf16_t* __restrict__ kT,
                            bf16_t* __restrict__ vW) {
  const int lane = threadIdx.x & 31, wave = threadIdx.x >> 5;
  const int n = lane & 15, hi = lane >> 4;
  const int t0 = blockIdx.x * 16;
  const int mc = blockIdx.y * 4 + wave;          // 0..23 chunks of 16 rows
  const int b  = blockIdx.z;

  const bf16_t* abase = Wcat + (size_t)mc * 16 * CIN;
  const bf16_t* bbase = xT + ((size_t)b * TQ + t0) * CIN;
  v8f acc = {};
#pragma unroll 4
  for (int k0 = 0; k0 < CIN; k0 += 32) {
    v16bf af = load_frag(abase, CIN, k0, lane);
    v16bf bf = load_frag(bbase, CIN, k0, lane);
    acc = wmma_bf16(af, bf, acc);
  }

  const int t = t0 + n;
  if (mc < 4) {                                   // q rows
    *(v8bf*)(qT + ((size_t)b * TQ + t) * DKEY + mc * 16 + 8 * hi) = pack8(acc);
    return;
  }
  // pooled outputs: max over t-pair (partner column = lane^1)
  float pr[8];
#pragma unroll
  for (int r = 0; r < 8; ++r)
    pr[r] = fmaxf(acc[r], __shfl_xor(acc[r], 1, 32));
  if (n & 1) return;
  const int s = (t0 >> 1) + (n >> 1);
  if (mc < 8) {                                   // k rows
    v8bf p;
#pragma unroll
    for (int r = 0; r < 8; ++r) p[r] = (bf16_t)pr[r];
    *(v8bf*)(kT + ((size_t)b * SK + s) * DKEY + (mc - 4) * 16 + 8 * hi) = p;
  } else {                                        // v rows, [cv][s] row-major
#pragma unroll
    for (int r = 0; r < 8; ++r) {
      const int cv = (mc - 8) * 16 + 8 * hi + r;
      vW[((size_t)b * DVAL + cv) * SK + s] = (bf16_t)pr[r];
    }
  }
}

// ---- Kernel D: fused attention -----------------------------------------
// One workgroup = one batch b x 16-query strip t0..t0+15, 256 threads.
// LDS: scores f32 [2048][16] (128KB) | attn^T bf16 [16][2048] (64KB)
//      | out^T bf16 [16][256] (8KB) | red f32[256] | colstat f32[32]
#define SC_OFF  0
#define AT_OFF  131072
#define OT_OFF  (131072 + 65536)
#define RD_OFF  (131072 + 65536 + 8192)
#define CS_OFF  (RD_OFF + 1024)
#define SMEM_BYTES (CS_OFF + 128)

__global__ void attn_kernel(const bf16_t* __restrict__ qT, const bf16_t* __restrict__ kT,
                            const bf16_t* __restrict__ vW, const bf16_t* __restrict__ WfB,
                            const float* __restrict__ x,  const float* __restrict__ gamma,
                            float* __restrict__ yOut, float* __restrict__ attnOut) {
  extern __shared__ char smem[];
  float*  scF   = (float*)(smem + SC_OFF);     // [s*16 + n]
  bf16_t* attnT = (bf16_t*)(smem + AT_OFF);    // [n*2048 + s]
  bf16_t* outT  = (bf16_t*)(smem + OT_OFF);    // [n*256 + cv]
  float*  red   = (float*)(smem + RD_OFF);     // [256]
  float*  cst   = (float*)(smem + CS_OFF);     // [0..15]=max [16..31]=1/sum

  const int tid = threadIdx.x, lane = tid & 31, wave = tid >> 5;
  const int n = lane & 15, hi = lane >> 4;
  const int t0 = blockIdx.x * 16, b = blockIdx.y;

  // ---- scores = k^T q : S[s,t] = sum_c kT[s,c]*qT[t,c] ----
  const bf16_t* qb = qT + ((size_t)b * TQ + t0) * DKEY;
  const v16bf qf0 = load_frag(qb, DKEY, 0, lane);
  const v16bf qf1 = load_frag(qb, DKEY, 32, lane);
  for (int sc = wave; sc < SK / 16; sc += 8) {
    const int s0 = sc * 16;
    const bf16_t* kb = kT + ((size_t)b * SK + s0) * DKEY;
    v8f acc = {};
    acc = wmma_bf16(load_frag(kb, DKEY, 0, lane), qf0, acc);
    acc = wmma_bf16(load_frag(kb, DKEY, 32, lane), qf1, acc);
#pragma unroll
    for (int r = 0; r < 8; ++r)
      scF[(s0 + 8 * hi + r) * 16 + n] = acc[r];   // SCALE==1
  }
  __syncthreads();

  // ---- softmax over s (column-wise), 16 cols x 16 partials ----
  const int col = tid & 15, part = tid >> 4;      // 128 rows per partial
  float pmax = -INFINITY;
  for (int i = 0; i < SK / 16; ++i)
    pmax = fmaxf(pmax, scF[(part * 128 + i) * 16 + col]);
  red[tid] = pmax;
  __syncthreads();
  if (tid < 16) {
    float m = red[tid];
#pragma unroll
    for (int p = 1; p < 16; ++p) m = fmaxf(m, red[p * 16 + tid]);
    cst[tid] = m;
  }
  __syncthreads();
  const float cmax = cst[col];
  float psum = 0.f;
  for (int i = 0; i < SK / 16; ++i) {
    const int idx = (part * 128 + i) * 16 + col;
    const float e = __expf(scF[idx] - cmax);
    scF[idx] = e;
    psum += e;
  }
  red[tid] = psum;
  __syncthreads();
  if (tid < 16) {
    float s = red[tid];
#pragma unroll
    for (int p = 1; p < 16; ++p) s += red[p * 16 + tid];
    cst[16 + tid] = 1.f / s;
  }
  __syncthreads();
  const float cinv = cst[16 + col];
  // normalize -> global attn (written exactly once) + bf16 transposed LDS copy
  for (int i = 0; i < SK / 16; ++i) {
    const int s = part * 128 + i;
    const float a = scF[s * 16 + col] * cinv;
    attnT[col * SK + s] = (bf16_t)a;
    attnOut[((size_t)b * SK + s) * TQ + t0 + col] = a;
  }
  __syncthreads();

  // ---- out = v @ attn : [256 x 16] strip, v streamed from L2 ----
  for (int mc = wave; mc < DVAL / 16; mc += 8) {
    const bf16_t* vb = vW + ((size_t)b * DVAL + mc * 16) * SK;
    v8f acc = {};
    for (int k0 = 0; k0 < SK; k0 += 32) {
      __builtin_prefetch(vb + (size_t)(lane & 15) * SK + k0 + 128, 0, 3);
      v16bf af = load_frag(vb, SK, k0, lane);
      v16bf bf = load_frag(attnT, SK, k0, lane);  // B^T[n][s] from LDS
      acc = wmma_bf16(af, bf, acc);
    }
    *(v8bf*)(outT + n * DVAL + mc * 16 + 8 * hi) = pack8(acc);
  }
  __syncthreads();

  // ---- y = gamma * (Wf @ out) + x ----
  const float g = gamma[0];
  for (int mc = wave; mc < CIN / 16; mc += 8) {
    const bf16_t* wb = WfB + (size_t)mc * 16 * DVAL;
    v8f acc = {};
#pragma unroll 2
    for (int k0 = 0; k0 < DVAL; k0 += 32) {
      v16bf af = load_frag(wb, DVAL, k0, lane);
      v16bf bf = load_frag(outT, DVAL, k0, lane); // B^T[n][cv] from LDS
      acc = wmma_bf16(af, bf, acc);
    }
#pragma unroll
    for (int r = 0; r < 8; ++r) {
      const int c = mc * 16 + 8 * hi + r;
      const size_t off = ((size_t)b * CIN + c) * TQ + t0 + n;
      yOut[off] = g * acc[r] + x[off];
    }
  }
}

// ---- launcher ----------------------------------------------------------
extern "C" void kernel_launch(void* const* d_in, const int* in_sizes, int n_in,
                              void* d_out, int out_size, void* d_ws, size_t ws_size,
                              hipStream_t stream) {
  const float* x     = (const float*)d_in[0];
  const float* gamma = (const float*)d_in[1];
  const float* Wq    = (const float*)d_in[2];
  const float* Wk    = (const float*)d_in[3];
  const float* Wv    = (const float*)d_in[4];
  const float* Wf    = (const float*)d_in[5];

  float* yOut    = (float*)d_out;                               // [8,512,4096]
  float* attnOut = yOut + (size_t)BATCH * CIN * TQ;             // [8,2048,4096]

  // workspace partition (~49 MB total)
  bf16_t* xT   = (bf16_t*)d_ws;                                 // [8,4096,512]
  bf16_t* Wcat = xT   + (size_t)BATCH * TQ * CIN;               // [384,512]
  bf16_t* WfB  = Wcat + (size_t)MPROJ * CIN;                    // [512,256]
  bf16_t* qT   = WfB  + (size_t)CIN * DVAL;                     // [8,4096,64]
  bf16_t* kT   = qT   + (size_t)BATCH * TQ * DKEY;              // [8,2048,64]
  bf16_t* vW   = kT   + (size_t)BATCH * SK * DKEY;              // [8,256,2048]

  prep_weights<<<1280, 256, 0, stream>>>(Wq, Wk, Wv, Wf, Wcat, WfB);
  transpose_x<<<dim3(TQ / 32, CIN / 32, BATCH), dim3(32, 8), 0, stream>>>(x, xT);
  proj_kernel<<<dim3(TQ / 16, MPROJ / 64, BATCH), 128, 0, stream>>>(Wcat, xT, qT, kT, vW);
  attn_kernel<<<dim3(TQ / 16, BATCH), 256, SMEM_BYTES, stream>>>(
      qT, kT, vW, WfB, x, gamma, yOut, attnOut);
}
